// SelfAttention_60017872994543
// MI455X (gfx1250) — compile-verified, matrix operand order checked by reading, not run on hardware
//
#include <hip/hip_runtime.h>
#include <hip/hip_bf16.h>
#include <math.h>

// ---------------------------------------------------------------------------
// SelfAttention block for MI455X (gfx1250, wave32, WMMA 16x16x32 f16)
//   x:[4,256,64,64] f32 -> GN -> q/k/v 1x1 -> softmax attn (N=4096) -> proj+res
// All GEMM stages run through v_wmma_f32_16x16x32_f16. Softmax is two-pass
// (row-stats then PV) so the 268MB attention matrix is never materialized.
// ---------------------------------------------------------------------------

typedef __attribute__((ext_vector_type(16))) _Float16 v16h;
typedef __attribute__((ext_vector_type(8)))  _Float16 v8h;
typedef __attribute__((ext_vector_type(8)))  float    v8f;

#define CC   256
#define GG   8
#define CG   32          // CC / GG
#define BB   4
#define NN   4096        // 64*64
#define EPSV 1e-5f

// ---- WMMA helpers ----------------------------------------------------------

static __device__ inline v8f wmma16(v16h a, v16h b, v8f c) {
  // D = A(16x32 f16) * B(32x16 f16) + C(16x16 f32)
  return __builtin_amdgcn_wmma_f32_16x16x32_f16(
      /*neg_a=*/false, a, /*neg_b=*/false, b,
      /*c_mod=*/(short)0, c, /*reuse_a=*/false, /*reuse_b=*/false);
}

// A-fragment loader for 16x32 f16 operand from a K-contiguous matrix
// (row-major [rows x K], stride ld in halves). Per the CDNA5 ISA layout:
//   lanes 0-15 : row = lane,    K in {0..7, 16..23}
//   lanes 16-31: row = lane-16, K in {8..15, 24..31}
// The same pattern serves the B operand when fed from B^T (n-major) storage.
static __device__ inline v16h load_frag_g(const _Float16* __restrict__ base, int ld,
                                          int row_base, int k_base, int lane) {
  int r    = row_base + (lane & 15);
  int koff = (lane >> 4) * 8;
  const _Float16* p = base + (long)r * ld + k_base + koff;
  v8h lo = *(const v8h*)(p);        // K = k_base+koff .. +7
  v8h hi = *(const v8h*)(p + 16);   // K = k_base+16+koff .. +7
  v16h out;
#pragma unroll
  for (int t = 0; t < 8; ++t) { out[t] = lo[t]; out[t + 8] = hi[t]; }
  return out;
}

// ---- Kernel 1: GroupNorm + transpose to hT[B,N,C] fp16 ---------------------

__global__ void gn_transpose_kernel(const float* __restrict__ x,
                                    const float* __restrict__ gw,
                                    const float* __restrict__ gb,
                                    _Float16* __restrict__ hT) {
  int g = blockIdx.x, b = blockIdx.y, tid = threadIdx.x;
  const float* xg = x + ((long)b * CC + (long)g * CG) * NN;

  float s = 0.f, s2 = 0.f;
  for (int f = tid; f < CG * NN; f += 256) {
    float v = xg[f]; s += v; s2 += v * v;
  }
  __shared__ float sh1[256], sh2[256];
  sh1[tid] = s; sh2[tid] = s2;
  __syncthreads();
  for (int off = 128; off > 0; off >>= 1) {
    if (tid < off) { sh1[tid] += sh1[tid + off]; sh2[tid] += sh2[tid + off]; }
    __syncthreads();
  }
  float inv  = 1.f / (float)(CG * NN);
  float mean = sh1[0] * inv;
  float var  = sh2[0] * inv - mean * mean;
  float rstd = rsqrtf(var + EPSV);

  int w = tid >> 5, lane = tid & 31;
  for (int c = w; c < CG; c += 8) {
    int ch = g * CG + c;
    float sc = gw[ch] * rstd;
    float of = gb[ch] - mean * sc;
    const float* row = xg + (long)c * NN;
    _Float16* dst = hT + (long)b * NN * CC + ch;
    for (int n = lane; n < NN; n += 32)
      dst[(long)n * CC] = (_Float16)(row[n] * sc + of);
  }
}

// ---- Kernel 2: fp32 -> fp16 weight convert ---------------------------------

__global__ void f32_to_f16_kernel(const float* __restrict__ src,
                                  _Float16* __restrict__ dst, int n) {
  int i = blockIdx.x * blockDim.x + threadIdx.x;
  if (i < n) dst[i] = (_Float16)src[i];
}

// ---- Kernel 3: generic WMMA GEMM  D[m,n] = sum_k A[m,k]*Bt[n,k] ------------
// 8 waves/block, one 16x16 tile per wave. fp16 out (+bias,*scale) or
// fp32 out (+bias + residual) for the final projection.

__global__ void gemm_wmma_kernel(const _Float16* __restrict__ A, int lda, long sA,
                                 const _Float16* __restrict__ Bt, int ldb, long sB,
                                 void* __restrict__ Dv, int ldd, long sD,
                                 const float* __restrict__ bias, int biasOnM,
                                 float scale,
                                 const float* __restrict__ resid, long sR, int K) {
  int wave = threadIdx.x >> 5, lane = threadIdx.x & 31;
  int m_base = (blockIdx.x * 8 + wave) * 16;
  int n_base = blockIdx.y * 16;
  int b = blockIdx.z;
  const _Float16* Ab = A + sA * b;
  const _Float16* Bb = Bt + sB * b;

  v8f acc = {0.f, 0.f, 0.f, 0.f, 0.f, 0.f, 0.f, 0.f};
  for (int k = 0; k < K; k += 32) {
    v16h af = load_frag_g(Ab, lda, m_base, k, lane);
    v16h bf = load_frag_g(Bb, ldb, n_base, k, lane);
    acc = wmma16(af, bf, acc);
  }

  int hi = (lane >= 16) ? 8 : 0;
  int n  = n_base + (lane & 15);
  if (resid) {
    float* Df = (float*)Dv + sD * b;
    const float* Rb = resid + sR * b;
#pragma unroll
    for (int r = 0; r < 8; ++r) {
      int m = m_base + r + hi;
      float vb = biasOnM ? bias[m] : bias[n];
      long idx = (long)m * ldd + n;
      Df[idx] = (acc[r] + vb) * scale + Rb[idx];
    }
  } else {
    _Float16* Dh = (_Float16*)Dv + sD * b;
#pragma unroll
    for (int r = 0; r < 8; ++r) {
      int m = m_base + r + hi;
      float vb = biasOnM ? bias[m] : bias[n];
      Dh[(long)m * ldd + n] = (_Float16)((acc[r] + vb) * scale);
    }
  }
}

// ---- Kernel 4: attention row stats (online max / sum-exp) ------------------
// S[i,j] = sum_c qT[i,c]*kT[j,c]  (1/sqrt(C) already folded into q)

__global__ void attn_rowstats_kernel(const _Float16* __restrict__ qT,
                                     const _Float16* __restrict__ kT,
                                     float* __restrict__ mrow,
                                     float* __restrict__ linv) {
  int wave = threadIdx.x >> 5, lane = threadIdx.x & 31;
  int b = blockIdx.y;
  int i_base = (blockIdx.x * 8 + wave) * 16;
  const _Float16* qb = qT + (long)b * NN * CC;
  const _Float16* kb = kT + (long)b * NN * CC;

  v16h qf[8];
#pragma unroll
  for (int kk = 0; kk < 8; ++kk) qf[kk] = load_frag_g(qb, CC, i_base, kk * 32, lane);

  float rm[8], rl[8];
#pragma unroll
  for (int r = 0; r < 8; ++r) { rm[r] = -3.0e38f; rl[r] = 0.f; }

  for (int jt = 0; jt < NN; jt += 16) {
    v8f s = {0.f, 0.f, 0.f, 0.f, 0.f, 0.f, 0.f, 0.f};
#pragma unroll
    for (int kk = 0; kk < 8; ++kk) {
      v16h bf = load_frag_g(kb, CC, jt, kk * 32, lane);
      s = wmma16(qf[kk], bf, s);
    }
    // D layout: vgpr r -> row (r + 8*(lane>=16)); col = lane&15.
#pragma unroll
    for (int r = 0; r < 8; ++r) {
      float v  = s[r];
      float tm = v;
#pragma unroll
      for (int msk = 1; msk < 16; msk <<= 1)
        tm = fmaxf(tm, __shfl_xor(tm, msk, 32));     // max over 16 cols (half-wave)
      float nm = fmaxf(rm[r], tm);
      float e  = __expf(v - nm);
#pragma unroll
      for (int msk = 1; msk < 16; msk <<= 1)
        e += __shfl_xor(e, msk, 32);                 // sum over 16 cols
      rl[r] = rl[r] * __expf(rm[r] - nm) + e;
      rm[r] = nm;
    }
  }
  if ((lane & 15) == 0) {
    int rowoff = (lane >= 16) ? 8 : 0;
#pragma unroll
    for (int r = 0; r < 8; ++r) {
      long idx = (long)b * NN + i_base + r + rowoff;
      mrow[idx] = rm[r];
      linv[idx] = 1.f / rl[r];
    }
  }
}

// ---- Kernel 5: PV — recompute S, P=exp(S-m)/l via LDS, O^T = P*V^T ---------

__global__ void attn_pv_kernel(const _Float16* __restrict__ qT,
                               const _Float16* __restrict__ kT,
                               const _Float16* __restrict__ vm,   // [B,C,N]
                               const float* __restrict__ mrow,
                               const float* __restrict__ linv,
                               _Float16* __restrict__ oT) {      // [B,N,C]
  __shared__ __align__(16) _Float16 sP[8][16 * 32];   // per-wave 16x32 P tile
  int wave = threadIdx.x >> 5, lane = threadIdx.x & 31;
  int b = blockIdx.y;
  int i_base = (blockIdx.x * 8 + wave) * 16;
  const _Float16* qb = qT + (long)b * NN * CC;
  const _Float16* kb = kT + (long)b * NN * CC;
  const _Float16* vb = vm + (long)b * CC * NN;
  int hi = (lane >= 16) ? 8 : 0;

  float mr[8], li[8];
#pragma unroll
  for (int r = 0; r < 8; ++r) {
    long idx = (long)b * NN + i_base + r + hi;
    mr[r] = mrow[idx]; li[r] = linv[idx];
  }

  v8f acc[16];
#pragma unroll
  for (int ct = 0; ct < 16; ++ct)
    acc[ct] = (v8f){0.f, 0.f, 0.f, 0.f, 0.f, 0.f, 0.f, 0.f};

  for (int j = 0; j < NN; j += 32) {
    v8f s0 = {0.f, 0.f, 0.f, 0.f, 0.f, 0.f, 0.f, 0.f};
    v8f s1 = {0.f, 0.f, 0.f, 0.f, 0.f, 0.f, 0.f, 0.f};
#pragma unroll
    for (int kk = 0; kk < 8; ++kk) {
      v16h aq = load_frag_g(qb, CC, i_base,  kk * 32, lane);
      v16h b0 = load_frag_g(kb, CC, j,       kk * 32, lane);
      v16h b1 = load_frag_g(kb, CC, j + 16,  kk * 32, lane);
      s0 = wmma16(aq, b0, s0);
      s1 = wmma16(aq, b1, s1);
    }
    // P = exp(S - m) / l, written to LDS row-major (16 x 32 over j)
    int n = lane & 15;
#pragma unroll
    for (int r = 0; r < 8; ++r) {
      int m = r + hi;
      sP[wave][m * 32 + n]      = (_Float16)(__expf(s0[r] - mr[r]) * li[r]);
      sP[wave][m * 32 + 16 + n] = (_Float16)(__expf(s1[r] - mr[r]) * li[r]);
    }
    asm volatile("s_wait_dscnt 0" ::: "memory");   // wave-local LDS RAW fence
    // Re-read P in the A-fragment split-K layout
    int arow = lane & 15, koff = (lane >> 4) * 8;
    v8h p0 = *(const v8h*)&sP[wave][arow * 32 + koff];
    v8h p1 = *(const v8h*)&sP[wave][arow * 32 + 16 + koff];
    v16h pA;
#pragma unroll
    for (int t = 0; t < 8; ++t) { pA[t] = p0[t]; pA[t + 8] = p1[t]; }
    // O^T[i, c] += P[i, j..j+31] * V^T[j..j+31, c]   (B operand from v[C,N])
#pragma unroll
    for (int ct = 0; ct < 16; ++ct) {
      v16h bf = load_frag_g(vb, NN, ct * 16, j, lane);
      acc[ct] = wmma16(pA, bf, acc[ct]);
    }
  }

  _Float16* ob = oT + (long)b * NN * CC;
#pragma unroll
  for (int ct = 0; ct < 16; ++ct) {
#pragma unroll
    for (int r = 0; r < 8; ++r) {
      int m = i_base + r + hi;
      ob[(long)m * CC + ct * 16 + (lane & 15)] = (_Float16)acc[ct][r];
    }
  }
}

// ---- Host launcher ---------------------------------------------------------

extern "C" void kernel_launch(void* const* d_in, const int* in_sizes, int n_in,
                              void* d_out, int out_size, void* d_ws, size_t ws_size,
                              hipStream_t stream) {
  (void)in_sizes; (void)n_in; (void)out_size; (void)ws_size;
  const float* x    = (const float*)d_in[0];
  const float* gn_w = (const float*)d_in[1];
  const float* gn_b = (const float*)d_in[2];
  const float* wq   = (const float*)d_in[3];
  const float* bq   = (const float*)d_in[4];
  const float* wk   = (const float*)d_in[5];
  const float* bk   = (const float*)d_in[6];
  const float* wv   = (const float*)d_in[7];
  const float* bv   = (const float*)d_in[8];
  const float* wp   = (const float*)d_in[9];
  const float* bp   = (const float*)d_in[10];
  float* out = (float*)d_out;

  // workspace carve-up (256B aligned)
  size_t off = 0;
  auto carve = [&](size_t bytes) {
    void* p = (char*)d_ws + off;
    off += (bytes + 255) & ~(size_t)255;
    return p;
  };
  const size_t actH = (size_t)BB * NN * CC * sizeof(_Float16);   // 8 MB each
  _Float16* hT   = (_Float16*)carve(actH);     // [B,N,C]
  _Float16* qT   = (_Float16*)carve(actH);     // [B,N,C] (scaled by 1/sqrt(C))
  _Float16* kT   = (_Float16*)carve(actH);     // [B,N,C]
  _Float16* vmat = (_Float16*)carve(actH);     // [B,C,N]
  _Float16* oT   = (_Float16*)carve(actH);     // [B,N,C]
  _Float16* wq16 = (_Float16*)carve((size_t)CC * CC * 2);
  _Float16* wk16 = (_Float16*)carve((size_t)CC * CC * 2);
  _Float16* wv16 = (_Float16*)carve((size_t)CC * CC * 2);
  _Float16* wp16 = (_Float16*)carve((size_t)CC * CC * 2);
  float* mrow = (float*)carve((size_t)BB * NN * 4);
  float* linv = (float*)carve((size_t)BB * NN * 4);

  // 1) GroupNorm -> hT fp16
  gn_transpose_kernel<<<dim3(GG, BB), 256, 0, stream>>>(x, gn_w, gn_b, hT);

  // 2) weights -> fp16
  f32_to_f16_kernel<<<CC * CC / 256, 256, 0, stream>>>(wq, wq16, CC * CC);
  f32_to_f16_kernel<<<CC * CC / 256, 256, 0, stream>>>(wk, wk16, CC * CC);
  f32_to_f16_kernel<<<CC * CC / 256, 256, 0, stream>>>(wv, wv16, CC * CC);
  f32_to_f16_kernel<<<CC * CC / 256, 256, 0, stream>>>(wp, wp16, CC * CC);

  const long sAct = (long)NN * CC;   // activation batch stride (elements)
  // 3) qT[i,o] = hT x wq^T, scale 1/sqrt(C)=1/16 folded in, +bq
  gemm_wmma_kernel<<<dim3(32, 16, BB), 256, 0, stream>>>(
      hT, CC, sAct, wq16, CC, 0, qT, CC, sAct, bq, 0, 0.0625f, nullptr, 0, CC);
  // 4) kT[j,o] = hT x wk^T, +bk
  gemm_wmma_kernel<<<dim3(32, 16, BB), 256, 0, stream>>>(
      hT, CC, sAct, wk16, CC, 0, kT, CC, sAct, bk, 0, 1.0f, nullptr, 0, CC);
  // 5) v[o,j] = wv x h, +bv (bias on M)
  gemm_wmma_kernel<<<dim3(2, 256, BB), 256, 0, stream>>>(
      wv16, CC, 0, hT, CC, sAct, vmat, NN, sAct, bv, 1, 1.0f, nullptr, 0, CC);

  // 6) softmax row stats
  attn_rowstats_kernel<<<dim3(32, BB), 256, 0, stream>>>(qT, kT, mrow, linv);
  // 7) O^T = softmax(S) * V^T
  attn_pv_kernel<<<dim3(32, BB), 256, 0, stream>>>(qT, kT, vmat, mrow, linv, oT);

  // 8) out = wp x O + bp + x  (fp32, bias on M, residual)
  gemm_wmma_kernel<<<dim3(2, 256, BB), 256, 0, stream>>>(
      wp16, CC, 0, oT, CC, sAct, out, NN, sAct, bp, 1, 1.0f, x, sAct, CC);
}
